// ResidualQuantizer_89283780149717
// MI455X (gfx1250) — matrix operand order repaired and checked by reading, not compiled, not run on hardware
//
#include <hip/hip_runtime.h>

#define KCODES 1024
#define DDIM   64
#define WAVES  8
#define ROWS_PER_WAVE 32
#define ROWS_PER_BLOCK (WAVES * ROWS_PER_WAVE)   // 256 rows / block

typedef __attribute__((ext_vector_type(16))) __bf16 v16bf;
typedef __attribute__((ext_vector_type(8)))  float  v8f;

// ws layout: [ float cnorm[3*1024] | __bf16 cb_hi[3*1024*64] | __bf16 cb_lo[3*1024*64] ]
#define WS_CN_BYTES   (3 * KCODES * 4)
#define WS_HI_OFF     (WS_CN_BYTES)
#define WS_CB_BYTES   (3 * KCODES * DDIM * 2)
#define WS_LO_OFF     (WS_HI_OFF + WS_CB_BYTES)
#define WS_NEED       (WS_LO_OFF + WS_CB_BYTES)

static __device__ __forceinline__ __bf16 f2bf(float f) { return (__bf16)f; }   // native v_cvt, RNE
static __device__ __forceinline__ float  bf2f(__bf16 b) { return (float)b; }

// Per-wave LDS ordering fence: DS ops are in-order per wave; this stops the
// compiler from reordering and drains DScnt before dependent cross-lane reads.
static __device__ __forceinline__ void lds_fence() {
  asm volatile("s_wait_dscnt 0" ::: "memory");
}

// Prep: per code row, compute ||c||^2 and bf16 hi/lo split of the codebook.
__global__ void rq_prep_kernel(const float* __restrict__ cb0,
                               const float* __restrict__ cb1,
                               const float* __restrict__ cb2,
                               char* __restrict__ ws) {
  int t = blockIdx.x * blockDim.x + threadIdx.x;   // 0..3071
  const float* cb = (t < KCODES) ? cb0 : ((t < 2 * KCODES) ? cb1 : cb2);
  int code = t & (KCODES - 1);
  const float* p = cb + (long)code * DDIM;

  float*  cn = (float*)ws;
  __bf16* hi = (__bf16*)(ws + WS_HI_OFF) + (long)t * DDIM;
  __bf16* lo = (__bf16*)(ws + WS_LO_OFF) + (long)t * DDIM;

  float s = 0.f;
#pragma unroll
  for (int d = 0; d < DDIM; ++d) {
    float v = p[d];
    __bf16 h = f2bf(v);
    hi[d] = h;
    lo[d] = f2bf(v - bf2f(h));
    s += v * v;
  }
  cn[t] = s;
}

template <bool PRE>
__global__ __launch_bounds__(256)
void rq_kernel(const float* __restrict__ latent,
               const float* __restrict__ cb0,
               const float* __restrict__ cb1,
               const float* __restrict__ cb2,
               const char* __restrict__ ws,
               float* __restrict__ out)
{
  __shared__ int   s_idx[WAVES][ROWS_PER_WAVE];     // 1 KB winner broadcast
  __shared__ float s_cn[PRE ? 16 : 3 * KCODES];     // norms only in fallback

  const int tid  = threadIdx.x;
  const int w    = tid >> 5;
  const int l    = tid & 31;
  const int arow = l & 15;          // WMMA A/B: lane -> M (or N)
  const int h    = l >> 4;          // k-half
  const long rowT0 = (long)blockIdx.x * ROWS_PER_BLOCK + (long)w * ROWS_PER_WAVE + arow;
  const long rowT1 = rowT0 + 16;

  if constexpr (!PRE) {
#pragma unroll
    for (int i = 0; i < 12; ++i) {
      int t2 = tid + i * 256;
      const float* cb = (t2 < KCODES) ? cb0 : ((t2 < 2 * KCODES) ? cb1 : cb2);
      const float* p = cb + (long)(t2 & (KCODES - 1)) * DDIM;
      float sacc = 0.f;
      for (int d = 0; d < DDIM; d += 4) {
        float4 q = *(const float4*)(p + d);
        sacc += q.x * q.x + q.y * q.y + q.z * q.z + q.w * q.w;
      }
      s_cn[t2] = sacc;
    }
    __syncthreads();
  }

  // Residual x lives ONLY as bf16 hi/lo A-fragments in VGPRs (WMMA A layout):
  // elems 0..15 = K c*32+16h+0..15
  v16bf a_hi[2][2], a_lo[2][2];   // [row-tile][k-chunk]
#pragma unroll
  for (int t = 0; t < 2; ++t) {
    const float* src = latent + (t == 0 ? rowT0 : rowT1) * DDIM;
#pragma unroll
    for (int c = 0; c < 2; ++c) {
#pragma unroll
      for (int part = 0; part < 2; ++part) {
        const float* p = src + c * 32 + h * 16 + part * 8;
        float4 q0 = *(const float4*)(p);
        float4 q1 = *(const float4*)(p + 4);
        float vv[8] = { q0.x, q0.y, q0.z, q0.w, q1.x, q1.y, q1.z, q1.w };
#pragma unroll
        for (int e = 0; e < 8; ++e) {
          __bf16 hh = f2bf(vv[e]);
          a_hi[t][c][part * 8 + e] = hh;
          a_lo[t][c][part * 8 + e] = f2bf(vv[e] - bf2f(hh));
        }
      }
    }
  }

  int sel_idx[3][2];   // winner code per stage per row-tile (row = arow)

#pragma unroll
  for (int s = 0; s < 3; ++s) {
    const float* cb = (s == 0) ? cb0 : ((s == 1) ? cb1 : cb2);
    const long   boff = (long)s * KCODES * DDIM;
    const __bf16* cbhi = (const __bf16*)(ws + WS_HI_OFF) + boff;
    const __bf16* cblo = (const __bf16*)(ws + WS_LO_OFF) + boff;
    const float*  cng  = (const float*)ws + s * KCODES;

    float vmin0[8], vmin1[8];
    int   vidx0[8], vidx1[8];
#pragma unroll
    for (int r = 0; r < 8; ++r) { vmin0[r] = vmin1[r] = 3.4e38f; vidx0[r] = vidx1[r] = 0; }

    // --- stream 64 column tiles of 16 codes; B shared by both row-tiles ---
#pragma unroll 2
    for (int kb = 0; kb < KCODES / 16; ++kb) {
      const int code = kb * 16 + arow;

      v16bf b_hi[2], b_lo[2];
      if constexpr (PRE) {
        const __bf16* bh = cbhi + (long)code * DDIM + h * 16;
        const __bf16* bl = cblo + (long)code * DDIM + h * 16;
        b_hi[0] = *(const v16bf*)(bh);
        b_hi[1] = *(const v16bf*)(bh + 32);
        b_lo[0] = *(const v16bf*)(bl);
        b_lo[1] = *(const v16bf*)(bl + 32);
      } else {
        const float* crow = cb + (long)code * DDIM;
#pragma unroll
        for (int c = 0; c < 2; ++c) {
          const float* p = crow + c * 32 + h * 16;
#pragma unroll
          for (int e = 0; e < 16; e += 4) {
            float4 q = *(const float4*)(p + e);
            float vv[4] = { q.x, q.y, q.z, q.w };
#pragma unroll
            for (int u = 0; u < 4; ++u) {
              __bf16 hh = f2bf(vv[u]);
              b_hi[c][e + u] = hh;
              b_lo[c][e + u] = f2bf(vv[u] - bf2f(hh));
            }
          }
        }
      }

      // dot = xh.ch + xl.ch + xh.cl per row-tile (bf16 hi/lo compensation)
      v8f acc0 = {}, acc1 = {};
      acc0 = __builtin_amdgcn_wmma_f32_16x16x32_bf16(false, a_hi[0][0], false, b_hi[0], (short)0, acc0, false, false);
      acc0 = __builtin_amdgcn_wmma_f32_16x16x32_bf16(false, a_hi[0][1], false, b_hi[1], (short)0, acc0, false, false);
      acc0 = __builtin_amdgcn_wmma_f32_16x16x32_bf16(false, a_lo[0][0], false, b_hi[0], (short)0, acc0, false, false);
      acc0 = __builtin_amdgcn_wmma_f32_16x16x32_bf16(false, a_hi[0][0], false, b_lo[0], (short)0, acc0, false, false);
      acc0 = __builtin_amdgcn_wmma_f32_16x16x32_bf16(false, a_lo[0][1], false, b_hi[1], (short)0, acc0, false, false);
      acc0 = __builtin_amdgcn_wmma_f32_16x16x32_bf16(false, a_hi[0][1], false, b_lo[1], (short)0, acc0, false, false);

      acc1 = __builtin_amdgcn_wmma_f32_16x16x32_bf16(false, a_hi[1][0], false, b_hi[0], (short)0, acc1, false, false);
      acc1 = __builtin_amdgcn_wmma_f32_16x16x32_bf16(false, a_hi[1][1], false, b_hi[1], (short)0, acc1, false, false);
      acc1 = __builtin_amdgcn_wmma_f32_16x16x32_bf16(false, a_lo[1][0], false, b_hi[0], (short)0, acc1, false, false);
      acc1 = __builtin_amdgcn_wmma_f32_16x16x32_bf16(false, a_hi[1][0], false, b_lo[0], (short)0, acc1, false, false);
      acc1 = __builtin_amdgcn_wmma_f32_16x16x32_bf16(false, a_lo[1][1], false, b_hi[1], (short)0, acc1, false, false);
      acc1 = __builtin_amdgcn_wmma_f32_16x16x32_bf16(false, a_hi[1][1], false, b_lo[1], (short)0, acc1, false, false);

      float cn;
      if constexpr (PRE) cn = cng[code];
      else               cn = s_cn[s * KCODES + code];

#pragma unroll
      for (int r = 0; r < 8; ++r) {
        float t0 = cn - 2.0f * acc0[r];
        if (t0 < vmin0[r]) { vmin0[r] = t0; vidx0[r] = code; }
        float t1 = cn - 2.0f * acc1[r];
        if (t1 < vmin1[r]) { vmin1[r] = t1; vidx1[r] = code; }
      }
    }

    // --- argmin across the 16 lanes of each half (rows 0-7 / 8-15) ---
#pragma unroll
    for (int m = 1; m < 16; m <<= 1) {
#pragma unroll
      for (int r = 0; r < 8; ++r) {
        float ov = __shfl_xor(vmin0[r], m, 32); int oi = __shfl_xor(vidx0[r], m, 32);
        if (ov < vmin0[r] || (ov == vmin0[r] && oi < vidx0[r])) { vmin0[r] = ov; vidx0[r] = oi; }
        float ow = __shfl_xor(vmin1[r], m, 32); int oj = __shfl_xor(vidx1[r], m, 32);
        if (ow < vmin1[r] || (ow == vmin1[r] && oj < vidx1[r])) { vmin1[r] = ow; vidx1[r] = oj; }
      }
    }
    if ((l & 15) == 0) {
      const int base = (l >> 4) * 8;
#pragma unroll
      for (int r = 0; r < 8; ++r) {
        s_idx[w][base + r]      = vidx0[r];
        s_idx[w][16 + base + r] = vidx1[r];
      }
    }
    lds_fence();
    sel_idx[s][0] = s_idx[w][arow];
    sel_idx[s][1] = s_idx[w][16 + arow];
    lds_fence();

    // --- residual update x <- x - q directly in A-fragment registers ---
    if (s < 2) {
#pragma unroll
      for (int t = 0; t < 2; ++t) {
        const float* qrow = cb + (long)sel_idx[s][t] * DDIM;
#pragma unroll
        for (int c = 0; c < 2; ++c) {
#pragma unroll
          for (int part = 0; part < 2; ++part) {
            const float* p = qrow + c * 32 + h * 16 + part * 8;
            float4 q0 = *(const float4*)(p);
            float4 q1 = *(const float4*)(p + 4);
            float qq[8] = { q0.x, q0.y, q0.z, q0.w, q1.x, q1.y, q1.z, q1.w };
#pragma unroll
            for (int e = 0; e < 8; ++e) {
              const int slot = part * 8 + e;
              float xv = bf2f(a_hi[t][c][slot]) + bf2f(a_lo[t][c][slot]) - qq[e];
              __bf16 hh = f2bf(xv);
              a_hi[t][c][slot] = hh;
              a_lo[t][c][slot] = f2bf(xv - bf2f(hh));
            }
          }
        }
      }
    }
  }

  // --- epilogue: out = ((x0 + x1) + x2) + q2, exact f32 chain (reference order),
  //     recomputed from L2-resident latent + the three winner rows ---
#pragma unroll
  for (int t = 0; t < 2; ++t) {
    const long   row = (t == 0 ? rowT0 : rowT1);
    const float* src = latent + row * DDIM;
    const float* q0r = cb0 + (long)sel_idx[0][t] * DDIM;
    const float* q1r = cb1 + (long)sel_idx[1][t] * DDIM;
    const float* q2r = cb2 + (long)sel_idx[2][t] * DDIM;
    float* orow = out + row * DDIM;
#pragma unroll
    for (int c = 0; c < 2; ++c) {
#pragma unroll
      for (int part = 0; part < 2; ++part) {
        const int dof = c * 32 + part * 16 + h * 8;
#pragma unroll
        for (int e = 0; e < 8; e += 4) {
          float4 x0 = *(const float4*)(src + dof + e);
          float4 q0 = *(const float4*)(q0r + dof + e);
          float4 q1 = *(const float4*)(q1r + dof + e);
          float4 q2 = *(const float4*)(q2r + dof + e);
          float4 o;
          { float x1 = x0.x - q0.x, x2 = x1 - q1.x; o.x = ((x0.x + x1) + x2) + q2.x; }
          { float x1 = x0.y - q0.y, x2 = x1 - q1.y; o.y = ((x0.y + x1) + x2) + q2.y; }
          { float x1 = x0.z - q0.z, x2 = x1 - q1.z; o.z = ((x0.z + x1) + x2) + q2.z; }
          { float x1 = x0.w - q0.w, x2 = x1 - q1.w; o.w = ((x0.w + x1) + x2) + q2.w; }
          *(float4*)(orow + dof + e) = o;
        }
      }
    }
  }
}

extern "C" void kernel_launch(void* const* d_in, const int* in_sizes, int n_in,
                              void* d_out, int out_size, void* d_ws, size_t ws_size,
                              hipStream_t stream) {
  const float* latent = (const float*)d_in[0];
  const float* cb0    = (const float*)d_in[1];
  const float* cb1    = (const float*)d_in[2];
  const float* cb2    = (const float*)d_in[3];
  float* out = (float*)d_out;
  char*  ws  = (char*)d_ws;

  const int nrows = in_sizes[0] / DDIM;            // 131072

  if (ws_size >= (size_t)WS_NEED) {
    rq_prep_kernel<<<(3 * KCODES) / 256, 256, 0, stream>>>(cb0, cb1, cb2, ws);
    rq_kernel<true><<<nrows / ROWS_PER_BLOCK, 256, 0, stream>>>(latent, cb0, cb1, cb2, ws, out);
  } else {
    // scratch too small: convert codebooks in-flight, norms computed per block
    rq_kernel<false><<<nrows / ROWS_PER_BLOCK, 256, 0, stream>>>(latent, cb0, cb1, cb2, ws, out);
  }
}
